// _MultiHeadScaledDotAttention_63359357551275
// MI455X (gfx1250) — matrix operand order repaired and checked by reading, compiled, not run on hardware
//
#include <hip/hip_runtime.h>

// MI455X (gfx1250) flash-attention + output projection.
// bf16 WMMA / f32 accum; row-sum via P*ones WMMA; row-max via DPP16 butterfly;
// K/V tiles fetched with async global->LDS (double buffered); mask bias
// precomputed into an LDS table; P/V key-index permuted for packed dword LDS.

typedef __attribute__((ext_vector_type(16))) __bf16 v16bf;
typedef __attribute__((ext_vector_type(8)))  float  v8f;

union fragu  { v16bf v; unsigned u[8]; };   // bf16 A/B fragment as 8 dwords
union bfpair { __bf16 h[2]; unsigned u; };  // packed bf16 pair

#define WMMA_BF16(Afr, Bfr, Cfr) \
  __builtin_amdgcn_wmma_f32_16x16x32_bf16(false, (Afr), false, (Bfr), (short)0, (Cfr), false, false)

static constexpr int Bc = 2;
static constexpr int Sc = 2048;
static constexpr int Dc = 1024;
static constexpr int Hc = 16;
static constexpr int Uc = 64;

// ---- feature probes --------------------------------------------------------
#if defined(__has_builtin)
#  if __has_builtin(__builtin_amdgcn_global_load_async_to_lds_b128)
#    define HAVE_ASYNC_LDS 1
#  endif
#  if __has_builtin(__builtin_amdgcn_update_dpp)
#    define HAVE_DPP 1
#  endif
#endif

#if defined(HAVE_ASYNC_LDS)
typedef int b128_t __attribute__((vector_size(16)));
#define ASYNC_B128(gp, lp)                                              \
  __builtin_amdgcn_global_load_async_to_lds_b128(                       \
      (__attribute__((address_space(1))) b128_t*)(gp),                  \
      (__attribute__((address_space(3))) b128_t*)(lp), 0, 0)
__device__ __forceinline__ void wait_async0() {
#  if __has_builtin(__builtin_amdgcn_s_wait_asynccnt)
  __builtin_amdgcn_s_wait_asynccnt(0);
#  else
  asm volatile("s_wait_asynccnt 0x0" ::: "memory");
#  endif
}
#endif

// 16-lane (DPP row) max butterfly: xor1, xor2 via quad_perm; +4, +8 via row_ror.
#if defined(HAVE_DPP)
template <int CTRL>
__device__ __forceinline__ float dpp_max_step(float x) {
  const int xi = __builtin_bit_cast(int, x);
  const int yi = __builtin_amdgcn_update_dpp(xi, xi, CTRL, 0xF, 0xF, true);
  return fmaxf(x, __builtin_bit_cast(float, yi));
}
__device__ __forceinline__ float rowmax16(float x) {
  x = dpp_max_step<0xB1>(x);    // quad_perm(1,0,3,2)  == xor 1
  x = dpp_max_step<0x4E>(x);    // quad_perm(2,3,0,1)  == xor 2
  x = dpp_max_step<0x124>(x);   // row_ror:4
  x = dpp_max_step<0x128>(x);   // row_ror:8
  return x;
}
#else
__device__ __forceinline__ float rowmax16(float x) {
  x = fmaxf(x, __shfl_xor(x, 1));
  x = fmaxf(x, __shfl_xor(x, 2));
  x = fmaxf(x, __shfl_xor(x, 4));
  x = fmaxf(x, __shfl_xor(x, 8));
  return x;
}
#endif

// ---------------------------------------------------------------------------
// Kernel 1: flash attention. One block = one (b,h) and 64 query rows.
// 4 waves x 16 rows, 32 keys per iteration.
// Key-index permutation: storage slot(j) = 2*(j%16) + (j/16), applied to BOTH
// P columns and V rows (PV contraction is permutation invariant).
// ---------------------------------------------------------------------------
__global__ __launch_bounds__(128)
void attn_fa_kernel(const float* __restrict__ q, const float* __restrict__ k,
                    const float* __restrict__ v, const float* __restrict__ mask,
                    float* __restrict__ attended)
{
    __shared__ __attribute__((aligned(16))) float stageK[2][32 * 64];
    __shared__ __attribute__((aligned(16))) float stageV[2][32 * 64];
    __shared__ __bf16 sK [32 * 66];        // bf16 K tile  [j_local][u]
    __shared__ __bf16 sVt[64 * 34];        // bf16 V tile  [u][slot]
    __shared__ __bf16 sP [4][16][36];      // per-wave P bounce [m][slot]
    __shared__ float  sBias[Sc];           // (1-mask)*1e4 for all key positions

    const int b   = blockIdx.z;
    const int h   = blockIdx.y;
    const int i0  = blockIdx.x * 64;
    const int tid = threadIdx.x;
    const int w   = tid >> 5;              // wave id (wave32)
    const int ln  = tid & 31;
    const int m   = ln & 15;
    const int hi  = ln >> 4;

    // ---- bias table: once per block
    #pragma unroll
    for (int e = 0; e < Sc / 128; ++e) {
        const int j = e * 128 + tid;
        sBias[j] = (1.0f - mask[b * Sc + j]) * 10000.0f;
    }

    // ---- Q as two A-fragments (u chunks 0..31 / 32..63), 1/sqrt(64) folded in
    fragu qa0, qa1;
    {
        const int qi = i0 + w * 16 + m;
        const float* qrow = q + (size_t)(b * Sc + qi) * Dc + h * Uc;
        #pragma unroll
        for (int g = 0; g < 8; ++g) {
            const int kb = 2 * g + 16 * hi;
            const float2 f0 = *(const float2*)(qrow + kb);
            const float2 f1 = *(const float2*)(qrow + 32 + kb);
            bfpair p0, p1;
            p0.h[0] = (__bf16)(f0.x * 0.125f);
            p0.h[1] = (__bf16)(f0.y * 0.125f);
            p1.h[0] = (__bf16)(f1.x * 0.125f);
            p1.h[1] = (__bf16)(f1.y * 0.125f);
            qa0.u[g] = p0.u;
            qa1.u[g] = p1.u;
        }
    }

    v16bf ones;
    #pragma unroll
    for (int i2 = 0; i2 < 16; ++i2) ones[i2] = (__bf16)1.0f;

    v8f o[4] = {v8f{}, v8f{}, v8f{}, v8f{}};   // O accum: 16 rows x 64 u
    v8f lacc = {};                             // row-sum accumulator (P * ones)
    float mrow[8];
    #pragma unroll
    for (int g = 0; g < 8; ++g) mrow[g] = -1e30f;

#if defined(HAVE_ASYNC_LDS)
    auto prefetch_tile = [&](int jn, int buf) {
        #pragma unroll
        for (int e = 0; e < 4; ++e) {          // 512 float4 per tensor / 128 thr
            const int id = e * 128 + tid;
            const int jl = id >> 4;
            const int uu = (id & 15) * 4;
            const size_t g0 = (size_t)(b * Sc + jn + jl) * Dc + h * Uc + uu;
            ASYNC_B128(k + g0, &stageK[buf][jl * 64 + uu]);
            ASYNC_B128(v + g0, &stageV[buf][jl * 64 + uu]);
        }
    };
    prefetch_tile(0, 0);
#endif

    for (int j0 = 0, it = 0; j0 < Sc; j0 += 32, ++it) {
        const int cur = it & 1;
        (void)cur;
#if defined(HAVE_ASYNC_LDS)
        wait_async0();
#endif
        __syncthreads();   // stage ready everywhere; previous tile fully consumed

        // ---- K: row-major packed dword stores (4 f32 -> 2 dwords per unit)
        #pragma unroll
        for (int e = 0; e < 4; ++e) {
            const int id = e * 128 + tid;
            const int jl = id >> 4;            // 0..31
            const int uu = (id & 15) * 4;      // 0..60
#if defined(HAVE_ASYNC_LDS)
            const float4 fk = *(const float4*)&stageK[cur][jl * 64 + uu];
#else
            const float4 fk = *(const float4*)(k + (size_t)(b * Sc + j0 + jl) * Dc + h * Uc + uu);
#endif
            bfpair p0, p1;
            p0.h[0] = (__bf16)fk.x; p0.h[1] = (__bf16)fk.y;
            p1.h[0] = (__bf16)fk.z; p1.h[1] = (__bf16)fk.w;
            *(unsigned*)&sK[jl * 66 + uu]     = p0.u;
            *(unsigned*)&sK[jl * 66 + uu + 2] = p1.u;
        }
        // ---- V: 2x2 logical block (rows x & x+16) -> packed slot pairs
        #pragma unroll
        for (int e = 0; e < 4; ++e) {
            const int id = e * 128 + tid;
            const int x  = id >> 5;            // 0..15
            const int uu = (id & 31) * 2;      // 0..62
#if defined(HAVE_ASYNC_LDS)
            const float2 f0 = *(const float2*)&stageV[cur][ x       * 64 + uu];
            const float2 f1 = *(const float2*)&stageV[cur][(x + 16) * 64 + uu];
#else
            const float2 f0 = *(const float2*)(v + (size_t)(b * Sc + j0 + x)      * Dc + h * Uc + uu);
            const float2 f1 = *(const float2*)(v + (size_t)(b * Sc + j0 + x + 16) * Dc + h * Uc + uu);
#endif
            bfpair pa2, pb2;
            pa2.h[0] = (__bf16)f0.x; pa2.h[1] = (__bf16)f1.x;   // slots 2x, 2x+1
            pb2.h[0] = (__bf16)f0.y; pb2.h[1] = (__bf16)f1.y;
            *(unsigned*)&sVt[ uu      * 34 + 2 * x] = pa2.u;
            *(unsigned*)&sVt[(uu + 1) * 34 + 2 * x] = pb2.u;
        }
        __syncthreads();

#if defined(HAVE_ASYNC_LDS)
        if (j0 + 32 < Sc) prefetch_tile(j0 + 32, 1 - cur);
#endif

        // ---- S = Q K^T : two 16x16 j-subtiles, 2 u-chunks each
        const float bias0 = sBias[j0 + m];
        const float bias1 = sBias[j0 + 16 + m];

        v8f s0 = {}, s1 = {};
        {
            fragu kb00, kb01, kb10, kb11;
            const int r0 = m * 66;
            const int r1 = (16 + m) * 66;
            #pragma unroll
            for (int g = 0; g < 8; ++g) {
                const int ku = 2 * g + 16 * hi;
                kb00.u[g] = *(const unsigned*)&sK[r0 + ku];
                kb01.u[g] = *(const unsigned*)&sK[r0 + 32 + ku];
                kb10.u[g] = *(const unsigned*)&sK[r1 + ku];
                kb11.u[g] = *(const unsigned*)&sK[r1 + 32 + ku];
            }
            s0 = WMMA_BF16(qa0.v, kb00.v, s0);
            s0 = WMMA_BF16(qa1.v, kb01.v, s0);
            s1 = WMMA_BF16(qa0.v, kb10.v, s1);
            s1 = WMMA_BF16(qa1.v, kb11.v, s1);
        }

        // ---- online softmax: max via DPP butterfly, sum via P*ones WMMA later
        float alpha[8];
        #pragma unroll
        for (int g = 0; g < 8; ++g) {
            const float a0 = s0[g] - bias0;
            const float a1 = s1[g] - bias1;
            const float t  = rowmax16(fmaxf(a0, a1));
            const float mn = fmaxf(mrow[g], t);
            alpha[g] = __expf(mrow[g] - mn);
            s0[g] = __expf(a0 - mn);
            s1[g] = __expf(a1 - mn);
            mrow[g] = mn;
        }
        #pragma unroll
        for (int t = 0; t < 4; ++t)
            #pragma unroll
            for (int g = 0; g < 8; ++g)
                o[t][g] *= alpha[g];
        #pragma unroll
        for (int g = 0; g < 8; ++g)
            lacc[g] *= alpha[g];

        // ---- P: C-layout -> A-layout via LDS bounce, packed dword per row
        //      (logical cols m, m+16 -> storage slots 2m, 2m+1)
        #pragma unroll
        for (int g = 0; g < 8; ++g) {
            bfpair pp;
            pp.h[0] = (__bf16)s0[g];
            pp.h[1] = (__bf16)s1[g];
            *(unsigned*)&sP[w][g + 8 * hi][2 * m] = pp.u;
        }
        __syncthreads();

        fragu pa;
        #pragma unroll
        for (int g = 0; g < 8; ++g) {
            const int kb = 2 * g + 16 * hi;
            pa.u[g] = *(const unsigned*)&sP[w][m][kb];
        }

        // ---- row sums: lacc += P * ones
        lacc = WMMA_BF16(pa.v, ones, lacc);

        // ---- O += P * V (4 u-chunk tiles, V rows in slot order)
        #pragma unroll
        for (int t = 0; t < 4; ++t) {
            fragu vb;
            const int ubase = (t * 16 + m) * 34;
            #pragma unroll
            for (int g = 0; g < 8; ++g) {
                const int kj = 2 * g + 16 * hi;
                vb.u[g] = *(const unsigned*)&sVt[ubase + kj];
            }
            o[t] = WMMA_BF16(pa.v, vb.v, o[t]);
        }
    }

    // ---- epilogue: attended[b, i, h*64 + u] = O / l
    #pragma unroll
    for (int g = 0; g < 8; ++g) {
        const float inv = 1.0f / lacc[g];      // row sum, replicated across lanes
        const int   row = i0 + w * 16 + g + 8 * hi;
        float* orow = attended + (size_t)(b * Sc + row) * Dc + h * Uc;
        orow[ 0 + m] = o[0][g] * inv;
        orow[16 + m] = o[1][g] * inv;
        orow[32 + m] = o[2][g] * inv;
        orow[48 + m] = o[3][g] * inv;
    }
}

// ---------------------------------------------------------------------------
// Kernel 2: out = attended @ W_o   (4096 x 1024 x 1024), bf16 WMMA tiles.
// 256 threads, 128x128 C tile, K staged 32 wide as packed bf16.
// ---------------------------------------------------------------------------
__global__ __launch_bounds__(256)
void proj_gemm_kernel(const float* __restrict__ a, const float* __restrict__ wgt,
                      float* __restrict__ out)
{
    __shared__ __bf16 sA[128 * 34];   // A tile [row][k]
    __shared__ __bf16 sB[128 * 34];   // B tile transposed [n][k]

    const int m0  = blockIdx.x * 128;
    const int n0  = blockIdx.y * 128;
    const int tid = threadIdx.x;
    const int w   = tid >> 5;
    const int ln  = tid & 31;
    const int m   = ln & 15;
    const int hi  = ln >> 4;

    v8f c[8] = {v8f{}, v8f{}, v8f{}, v8f{}, v8f{}, v8f{}, v8f{}, v8f{}};

    for (int k0 = 0; k0 < Dc; k0 += 32) {
        __syncthreads();
        #pragma unroll
        for (int e = 0; e < 8; ++e) {
            const int id = e * 256 + tid;            // 2048 f32-pairs each
            const int r  = id >> 4;
            const int kk = (id & 15) * 2;
            const float2 fa = *(const float2*)(a + (size_t)(m0 + r) * Dc + k0 + kk);
            bfpair pk;
            pk.h[0] = (__bf16)fa.x; pk.h[1] = (__bf16)fa.y;
            *(unsigned*)&sA[r * 34 + kk] = pk.u;

            const int k2 = id >> 6;
            const int nn = (id & 63) * 2;
            const float2 fb = *(const float2*)(wgt + (size_t)(k0 + k2) * Dc + n0 + nn);
            sB[ nn      * 34 + k2] = (__bf16)fb.x;
            sB[(nn + 1) * 34 + k2] = (__bf16)fb.y;
        }
        __syncthreads();

        fragu af;
        const int abase = (w * 16 + m) * 34;
        #pragma unroll
        for (int g = 0; g < 8; ++g) {
            const int kb = 2 * g + 16 * hi;
            af.u[g] = *(const unsigned*)&sA[abase + kb];
        }
        #pragma unroll
        for (int tn = 0; tn < 8; ++tn) {
            fragu bfrag;
            const int cbase = (tn * 16 + m) * 34;
            #pragma unroll
            for (int g = 0; g < 8; ++g) {
                const int kk = 2 * g + 16 * hi;
                bfrag.u[g] = *(const unsigned*)&sB[cbase + kk];
            }
            c[tn] = WMMA_BF16(af.v, bfrag.v, c[tn]);
        }
    }

    #pragma unroll
    for (int tn = 0; tn < 8; ++tn)
        #pragma unroll
        for (int g = 0; g < 8; ++g)
            out[(size_t)(m0 + w * 16 + g + 8 * hi) * Dc + n0 + tn * 16 + m] = c[tn][g];
}

// ---------------------------------------------------------------------------
extern "C" void kernel_launch(void* const* d_in, const int* in_sizes, int n_in,
                              void* d_out, int out_size, void* d_ws, size_t ws_size,
                              hipStream_t stream) {
    (void)in_sizes; (void)n_in; (void)out_size; (void)ws_size;
    const float* q    = (const float*)d_in[0];
    const float* k    = (const float*)d_in[1];
    const float* v    = (const float*)d_in[2];
    const float* mask = (const float*)d_in[3];
    const float* wo   = (const float*)d_in[4];
    float* out        = (float*)d_out;
    float* attended   = (float*)d_ws;   // B*S*D f32 = 16 MB scratch

    dim3 g1(Sc / 64, Hc, Bc), b1(128);
    attn_fa_kernel<<<g1, b1, 0, stream>>>(q, k, v, mask, attended);

    dim3 g2((Bc * Sc) / 128, Dc / 128), b2(256);
    proj_gemm_kernel<<<g2, b2, 0, stream>>>(attended, wo, out);
}
